// ResidualVQ_45148696216527
// MI455X (gfx1250) — matrix operand order repaired and checked by reading, not compiled, not run on hardware
//
#include <hip/hip_runtime.h>
#include <hip/hip_bf16.h>
#include <stdint.h>

// ---------------------------------------------------------------------------
// ResidualVQ for MI455X (gfx1250, wave32, WMMA).
// Pipeline:
//   P1: embedding fp32 -> eb bf16 [K,D] + e_norm fp32 [K]
//   P2: x         fp32 -> xb bf16 [N,D] + x_norm fp32 [N]
//   P3: W1,W2     fp32 -> W?T bf16 [E,D] (transposed for contiguous B-frags)
//   B : distance GEMM via v_wmma_f32_16x16x32_bf16 + in-register argmin,
//       K split in 2 chunks; B-tile double-buffered in registers so the
//       8-WMMA accumulate chain overlaps the next tile's global_load_b128s
//   M : merge 2 chunk candidates, add ||x||^2
//   C : gather z, r = x - z, h = relu(r@W1+b1), out = z + h@W2 + b2 (2x WMMA)
//   L : loss = BETA * mean(min_dist)
// ---------------------------------------------------------------------------

typedef __attribute__((ext_vector_type(16))) __bf16 v16bf;
typedef __attribute__((ext_vector_type(8)))  float  v8f;

struct __align__(16) B128 { uint32_t w[4]; };

union Frag16 {
  uint32_t u[8];
  v16bf    v;
};

__device__ __forceinline__ unsigned short f32_to_bf16(float f) {
  union { float f; uint32_t u; } c;
  c.f = f;
  uint32_t u = c.u;
  u += 0x7FFFu + ((u >> 16) & 1u);   // round to nearest even
  return (unsigned short)(u >> 16);
}

#define N_TOK 16384
#define DIM   256
#define KCODE 8192
#define KCHUNK 4096
#define BETA  0.25f

// --------------------------------------------------------------------------
// P1/P2: fp32 [R,256] -> bf16 [R,256], row squared norm -> norm[R]
// --------------------------------------------------------------------------
__global__ void __launch_bounds__(256)
conv_norm_kernel(const float* __restrict__ src, unsigned short* __restrict__ dst,
                 float* __restrict__ norm) {
  const int row = blockIdx.x;
  const int d   = threadIdx.x;
  float v = src[(size_t)row * DIM + d];
  dst[(size_t)row * DIM + d] = f32_to_bf16(v);
  float s = v * v;
  #pragma unroll
  for (int off = 16; off >= 1; off >>= 1) s += __shfl_xor(s, off, 32);
  __shared__ float red[8];
  if ((threadIdx.x & 31) == 0) red[threadIdx.x >> 5] = s;
  __syncthreads();
  if (threadIdx.x == 0) {
    float t = 0.f;
    #pragma unroll
    for (int i = 0; i < 8; ++i) t += red[i];
    norm[row] = t;
  }
}

// --------------------------------------------------------------------------
// P3: W [D,E] fp32 -> WT [E,D] bf16
// --------------------------------------------------------------------------
__global__ void __launch_bounds__(256)
wtrans_kernel(const float* __restrict__ w, unsigned short* __restrict__ wt) {
  const int d = blockIdx.x;     // input-feature row of W
  const int e = threadIdx.x;    // output feature
  wt[(size_t)e * DIM + d] = f32_to_bf16(w[(size_t)d * DIM + e]);
}

// --------------------------------------------------------------------------
// B: distance argmin. Block = 8 waves x 16 rows = 128 rows. grid = (128, 2).
// Wave tile: 16 rows x 16 codes, full D=256 in registers (8 A-frags).
// score(k) = ||e_k||^2 - 2 * x.e   (||x||^2 added at merge)
// B tiles double-buffered: loads for tile kt+16 issue before the WMMA chain
// of tile kt, so s_wait_loadcnt overlaps ~8 WMMAs of work.
// --------------------------------------------------------------------------
__global__ void __launch_bounds__(256)
vq_argmin_kernel(const unsigned short* __restrict__ xb,
                 const unsigned short* __restrict__ eb,
                 const float* __restrict__ e_norm,
                 float* __restrict__ md_part,   // [2, N]
                 int*   __restrict__ idx_part)  // [2, N]
{
  const int lane = threadIdx.x & 31;
  const int wave = threadIdx.x >> 5;
  const int half = lane >> 4;     // which 16-lane group
  const int lm   = lane & 15;
  const int rowbase = blockIdx.x * 128 + wave * 16;
  const int rowA    = rowbase + lm;
  const int kc      = blockIdx.y;           // K chunk
  const int k0      = kc * KCHUNK;

  // A-fragments for the whole D=256 reduction: 8 frags = 64 VGPRs.
  // Layout (16-bit A 16x32): lane(half,m) holds row m, d in
  //   [dt + 8*half, +8) and [dt + 16 + 8*half, +8)  -> two contiguous b128.
  Frag16 a[8];
  #pragma unroll
  for (int t = 0; t < 8; ++t) {
    const unsigned short* p = xb + (size_t)rowA * DIM + t * 32 + half * 8;
    B128 lo = *(const B128*)p;
    B128 hi = *(const B128*)(p + 16);
    a[t].u[0] = lo.w[0]; a[t].u[1] = lo.w[1]; a[t].u[2] = lo.w[2]; a[t].u[3] = lo.w[3];
    a[t].u[4] = hi.w[0]; a[t].u[5] = hi.w[1]; a[t].u[6] = hi.w[2]; a[t].u[7] = hi.w[3];
  }

  // B-fragment loader: lane(half,n) holds column n = code kt+lm, rows
  // d in [dt + 8*half, +8) and [dt + 16 + 8*half, +8) of codebook row kt+lm.
  const unsigned short* ebL = eb + (size_t)lm * DIM + 8 * half;
  auto load_btile = [&](Frag16* b, int kt) {
    const unsigned short* brow = ebL + (size_t)kt * DIM;
    #pragma unroll
    for (int t = 0; t < 8; ++t) {
      B128 lo = *(const B128*)(brow + t * 32);
      B128 hi = *(const B128*)(brow + t * 32 + 16);
      b[t].u[0] = lo.w[0]; b[t].u[1] = lo.w[1]; b[t].u[2] = lo.w[2]; b[t].u[3] = lo.w[3];
      b[t].u[4] = hi.w[0]; b[t].u[5] = hi.w[1]; b[t].u[6] = hi.w[2]; b[t].u[7] = hi.w[3];
    }
  };

  float best[8];
  int   besti[8];
  #pragma unroll
  for (int j = 0; j < 8; ++j) { best[j] = 3.4e38f; besti[j] = 0; }

  Frag16 b0[8], b1[8];
  load_btile(b0, k0);

  for (int kt = k0; kt < k0 + KCHUNK; kt += 32) {
    const float en0 = e_norm[kt + lm];
    const float en1 = e_norm[kt + 16 + lm];

    // Issue next sub-tile's loads before the WMMA chain on b0.
    load_btile(b1, kt + 16);
    {
      v8f acc = {};
      #pragma unroll
      for (int t = 0; t < 8; ++t)
        acc = __builtin_amdgcn_wmma_f32_16x16x32_bf16(
            false, a[t].v, false, b0[t].v, (short)0, acc, false, false);
      #pragma unroll
      for (int j = 0; j < 8; ++j) {
        float score = en0 - 2.0f * acc[j];
        if (score < best[j]) { best[j] = score; besti[j] = kt + lm; }
      }
    }

    // Loads for kt+32 (tail read covered by 16 padding rows after eb).
    load_btile(b0, kt + 32);
    {
      v8f acc = {};
      #pragma unroll
      for (int t = 0; t < 8; ++t)
        acc = __builtin_amdgcn_wmma_f32_16x16x32_bf16(
            false, a[t].v, false, b1[t].v, (short)0, acc, false, false);
      #pragma unroll
      for (int j = 0; j < 8; ++j) {
        float score = en1 - 2.0f * acc[j];
        if (score < best[j]) { best[j] = score; besti[j] = kt + 16 + lm; }
      }
    }
  }

  // Cross-lane argmin inside each 16-lane column group (xor stays in group).
  #pragma unroll
  for (int j = 0; j < 8; ++j) {
    float b  = best[j];
    int   bi = besti[j];
    #pragma unroll
    for (int off = 8; off >= 1; off >>= 1) {
      float ob  = __shfl_xor(b, off, 32);
      int   obi = __shfl_xor(bi, off, 32);
      if (ob < b || (ob == b && obi < bi)) { b = ob; bi = obi; }
    }
    if (lm == 0) {
      const int row = rowbase + j + 8 * half;   // C-matrix row mapping
      md_part[(size_t)kc * N_TOK + row]  = b;
      idx_part[(size_t)kc * N_TOK + row] = bi;
    }
  }
}

// --------------------------------------------------------------------------
// M: merge the two K-chunk candidates; add ||x||^2.
// --------------------------------------------------------------------------
__global__ void __launch_bounds__(256)
merge_kernel(const float* __restrict__ md_part, const int* __restrict__ idx_part,
             const float* __restrict__ x_norm,
             float* __restrict__ min_dist, int* __restrict__ idx) {
  const int n = blockIdx.x * 256 + threadIdx.x;
  float s0 = md_part[n], s1 = md_part[N_TOK + n];
  int   i0 = idx_part[n], i1 = idx_part[N_TOK + n];
  bool take1 = s1 < s0;   // chunk-1 indices are larger: tie keeps chunk 0
  min_dist[n] = (take1 ? s1 : s0) + x_norm[n];
  idx[n]      = take1 ? i1 : i0;
}

// --------------------------------------------------------------------------
// C: gather + residual MLP. Block = 512 threads = 16 waves, 16 rows.
// Wave w owns output features [16w, 16w+16).
// --------------------------------------------------------------------------
#define RS 264   // bf16 LDS row stride (528 B: 16B-aligned, bank-staggered)
#define ZS 260   // fp32 LDS row stride

__global__ void __launch_bounds__(512)
vq_mlp_kernel(const float* __restrict__ x, const float* __restrict__ emb,
              const int* __restrict__ idx,
              const unsigned short* __restrict__ w1t, const float* __restrict__ b1,
              const unsigned short* __restrict__ w2t, const float* __restrict__ b2,
              float* __restrict__ out) {
  __shared__ __attribute__((aligned(16))) float          zs[16 * ZS];
  __shared__ __attribute__((aligned(16))) unsigned short rs[16 * RS];
  __shared__ __attribute__((aligned(16))) unsigned short hs[16 * RS];

  const int tid  = threadIdx.x;
  const int lane = tid & 31;
  const int wave = tid >> 5;
  const int half = lane >> 4;
  const int lm   = lane & 15;
  const int rowbase = blockIdx.x * 16;

  // Phase 1: z = emb[idx], r = x - z (bf16), both staged in LDS.
  for (int i = tid; i < 16 * DIM; i += 512) {
    const int row = i >> 8, d = i & 255;
    const int n = rowbase + row;
    const int code = idx[n];
    float zv = emb[(size_t)code * DIM + d];
    float xv = x[(size_t)n * DIM + d];
    zs[row * ZS + d] = zv;
    rs[row * RS + d] = f32_to_bf16(xv - zv);
  }
  __syncthreads();

  const int col = wave * 16 + lm;   // global output feature for this lane

  // Phase 2: h = relu(r @ W1 + b1)
  {
    v8f acc = {};
    #pragma unroll
    for (int t = 0; t < 8; ++t) {
      Frag16 a, b;
      const unsigned short* ap = &rs[lm * RS + t * 32 + half * 8];
      B128 alo = *(const B128*)ap;
      B128 ahi = *(const B128*)(ap + 16);
      a.u[0] = alo.w[0]; a.u[1] = alo.w[1]; a.u[2] = alo.w[2]; a.u[3] = alo.w[3];
      a.u[4] = ahi.w[0]; a.u[5] = ahi.w[1]; a.u[6] = ahi.w[2]; a.u[7] = ahi.w[3];
      const unsigned short* bp = w1t + (size_t)col * DIM + t * 32 + 8 * half;
      B128 blo = *(const B128*)bp;
      B128 bhi = *(const B128*)(bp + 16);
      b.u[0] = blo.w[0]; b.u[1] = blo.w[1]; b.u[2] = blo.w[2]; b.u[3] = blo.w[3];
      b.u[4] = bhi.w[0]; b.u[5] = bhi.w[1]; b.u[6] = bhi.w[2]; b.u[7] = bhi.w[3];
      acc = __builtin_amdgcn_wmma_f32_16x16x32_bf16(
          false, a.v, false, b.v, (short)0, acc, false, false);
    }
    const float bias = b1[col];
    #pragma unroll
    for (int j = 0; j < 8; ++j) {
      float h = acc[j] + bias;
      h = h > 0.f ? h : 0.f;
      hs[(j + 8 * half) * RS + col] = f32_to_bf16(h);
    }
  }
  __syncthreads();

  // Phase 3: out = z + h @ W2 + b2
  {
    v8f acc = {};
    #pragma unroll
    for (int t = 0; t < 8; ++t) {
      Frag16 a, b;
      const unsigned short* ap = &hs[lm * RS + t * 32 + half * 8];
      B128 alo = *(const B128*)ap;
      B128 ahi = *(const B128*)(ap + 16);
      a.u[0] = alo.w[0]; a.u[1] = alo.w[1]; a.u[2] = alo.w[2]; a.u[3] = alo.w[3];
      a.u[4] = ahi.w[0]; a.u[5] = ahi.w[1]; a.u[6] = ahi.w[2]; a.u[7] = ahi.w[3];
      const unsigned short* bp = w2t + (size_t)col * DIM + t * 32 + 8 * half;
      B128 blo = *(const B128*)bp;
      B128 bhi = *(const B128*)(bp + 16);
      b.u[0] = blo.w[0]; b.u[1] = blo.w[1]; b.u[2] = blo.w[2]; b.u[3] = blo.w[3];
      b.u[4] = bhi.w[0]; b.u[5] = bhi.w[1]; b.u[6] = bhi.w[2]; b.u[7] = bhi.w[3];
      acc = __builtin_amdgcn_wmma_f32_16x16x32_bf16(
          false, a.v, false, b.v, (short)0, acc, false, false);
    }
    const float bias = b2[col];
    #pragma unroll
    for (int j = 0; j < 8; ++j) {
      const int row = j + 8 * half;
      float r = acc[j] + bias + zs[row * ZS + col];
      out[(size_t)(rowbase + row) * DIM + col] = r;
    }
  }
}

// --------------------------------------------------------------------------
// L: loss = BETA * mean(min_dist). One block, 1024 threads.
// --------------------------------------------------------------------------
__global__ void __launch_bounds__(1024)
loss_kernel(const float* __restrict__ min_dist, float* __restrict__ out_loss) {
  float s = 0.f;
  for (int i = threadIdx.x; i < N_TOK; i += 1024) s += min_dist[i];
  #pragma unroll
  for (int off = 16; off >= 1; off >>= 1) s += __shfl_xor(s, off, 32);
  __shared__ float red[32];
  if ((threadIdx.x & 31) == 0) red[threadIdx.x >> 5] = s;
  __syncthreads();
  if (threadIdx.x < 32) {
    float t = red[threadIdx.x];
    #pragma unroll
    for (int off = 16; off >= 1; off >>= 1) t += __shfl_xor(t, off, 32);
    if (threadIdx.x == 0) *out_loss = BETA * t / (float)N_TOK;
  }
}

// --------------------------------------------------------------------------
// Workspace layout (bytes). eb has 16 padding rows (8 KB) so the double
// buffer's tail load at kt+32 == chunk end stays in mapped memory.
// --------------------------------------------------------------------------
#define OFF_EB    ((size_t)0)           // 8192*256*2 + 16*256*2 = 4202496
#define OFF_XB    ((size_t)4202496)     // 16384*256*2 = 8388608
#define OFF_ENORM ((size_t)12591104)    // 8192*4
#define OFF_XNORM ((size_t)12623872)    // 16384*4
#define OFF_IDX   ((size_t)12689408)    // 16384*4
#define OFF_MD    ((size_t)12754944)    // 16384*4
#define OFF_W1T   ((size_t)12820480)    // 256*256*2
#define OFF_W2T   ((size_t)12951552)    // 256*256*2
#define OFF_MDP   ((size_t)13082624)    // 2*16384*4
#define OFF_IDXP  ((size_t)13213696)    // 2*16384*4

extern "C" void kernel_launch(void* const* d_in, const int* in_sizes, int n_in,
                              void* d_out, int out_size, void* d_ws, size_t ws_size,
                              hipStream_t stream) {
  (void)in_sizes; (void)n_in; (void)ws_size;
  const float* x   = (const float*)d_in[0];   // [B,S,D]
  const float* emb = (const float*)d_in[1];   // [K,D]
  const float* W1  = (const float*)d_in[2];   // [D,D]
  const float* b1  = (const float*)d_in[3];   // [D]
  const float* W2  = (const float*)d_in[4];   // [D,D]
  const float* b2  = (const float*)d_in[5];   // [D]
  float* out = (float*)d_out;

  char* ws = (char*)d_ws;
  unsigned short* eb     = (unsigned short*)(ws + OFF_EB);
  unsigned short* xb     = (unsigned short*)(ws + OFF_XB);
  float*          e_norm = (float*)(ws + OFF_ENORM);
  float*          x_norm = (float*)(ws + OFF_XNORM);
  int*            idx    = (int*)(ws + OFF_IDX);
  float*          md     = (float*)(ws + OFF_MD);
  unsigned short* w1t    = (unsigned short*)(ws + OFF_W1T);
  unsigned short* w2t    = (unsigned short*)(ws + OFF_W2T);
  float*          mdp    = (float*)(ws + OFF_MDP);
  int*            idxp   = (int*)(ws + OFF_IDXP);

  // Precision prep (one-time per call; ~12 MB, trivially bandwidth bound)
  conv_norm_kernel<<<KCODE, 256, 0, stream>>>(emb, eb, e_norm);
  conv_norm_kernel<<<N_TOK, 256, 0, stream>>>(x, xb, x_norm);
  wtrans_kernel<<<DIM, 256, 0, stream>>>(W1, w1t);
  wtrans_kernel<<<DIM, 256, 0, stream>>>(W2, w2t);

  // Distance GEMM + argmin (69 GFLOP on the bf16 WMMA pipes)
  dim3 gB(N_TOK / 128, 2);
  vq_argmin_kernel<<<gB, 256, 0, stream>>>(xb, eb, e_norm, mdp, idxp);
  merge_kernel<<<N_TOK / 256, 256, 0, stream>>>(mdp, idxp, x_norm, md, idx);

  // Gather + residual MLP (2x WMMA GEMMs per tile)
  vq_mlp_kernel<<<N_TOK / 16, 512, 0, stream>>>(x, emb, idx, w1t, b1, w2t, b2, out);

  // Loss scalar -> last element of d_out
  loss_kernel<<<1, 1024, 0, stream>>>(md, out + (size_t)out_size - 1);
}
